// EncoderLayer_65103114273211
// MI455X (gfx1250) — matrix operand order repaired and checked by reading, not compiled, not run on hardware
//
#include <hip/hip_runtime.h>
#include <hip/hip_bf16.h>

typedef __attribute__((ext_vector_type(16))) __bf16 v16bf;
typedef __attribute__((ext_vector_type(8)))  __bf16 v8bf;
typedef __attribute__((ext_vector_type(8)))  float  v8f;

// ---------------------------------------------------------------- helpers

__device__ __forceinline__ __bf16 tobf(float f) {
    unsigned int u = __builtin_bit_cast(unsigned int, f);
    u += 0x7fffu + ((u >> 16) & 1u);              // round-to-nearest-even
    unsigned short h = (unsigned short)(u >> 16);
    return __builtin_bit_cast(__bf16, h);
}

__device__ __forceinline__ v16bf combine8(v8bf lo, v8bf hi) {
    return __builtin_shufflevector(lo, hi, 0,1,2,3,4,5,6,7,8,9,10,11,12,13,14,15);
}

__device__ __forceinline__ v8f wmma_bf16(v16bf a, v16bf b, v8f c) {
    // (neg_a, A, neg_b, B, c_mod, C, reuse_a, reuse_b)
    return __builtin_amdgcn_wmma_f32_16x16x32_bf16(false, a, false, b, (short)0, c,
                                                   false, false);
}

// ---------------------------------------------------------------- LayerNorm -> bf16

__global__ __launch_bounds__(256) void ln_bf16_kernel(
    const float* __restrict__ x, const float* __restrict__ g,
    const float* __restrict__ bb, __bf16* __restrict__ out)
{
    const int row = blockIdx.x;
    const float* xr = x + (size_t)row * 1024;
    const int t = threadIdx.x;
    float v0 = xr[t], v1 = xr[t + 256], v2 = xr[t + 512], v3 = xr[t + 768];
    float s  = v0 + v1 + v2 + v3;
    float ss = v0*v0 + v1*v1 + v2*v2 + v3*v3;
    #pragma unroll
    for (int off = 16; off; off >>= 1) {
        s  += __shfl_xor(s,  off, 32);
        ss += __shfl_xor(ss, off, 32);
    }
    __shared__ float rs[8], rss[8];
    const int wave = t >> 5, lane = t & 31;
    if (lane == 0) { rs[wave] = s; rss[wave] = ss; }
    __syncthreads();
    s = 0.f; ss = 0.f;
    #pragma unroll
    for (int i = 0; i < 8; ++i) { s += rs[i]; ss += rss[i]; }
    const float mu   = s * (1.0f / 1024.0f);
    const float var  = ss * (1.0f / 1024.0f) - mu * mu;
    const float rstd = rsqrtf(var + 1e-5f);
    __bf16* orow = out + (size_t)row * 1024;
    orow[t      ] = tobf((v0 - mu) * rstd * g[t      ] + bb[t      ]);
    orow[t + 256] = tobf((v1 - mu) * rstd * g[t + 256] + bb[t + 256]);
    orow[t + 512] = tobf((v2 - mu) * rstd * g[t + 512] + bb[t + 512]);
    orow[t + 768] = tobf((v3 - mu) * rstd * g[t + 768] + bb[t + 768]);
}

// ---------------------------------------------------------------- weight conversion

// Wq/Wk/Wv are [H=16, D=1024, DK=64] -> packed bf16 [D=1024, 3072] (Q|K|V columns)
__global__ __launch_bounds__(256) void convert_qkv_w(
    const float* __restrict__ Wq, const float* __restrict__ Wk,
    const float* __restrict__ Wv, __bf16* __restrict__ Wqkv)
{
    const int idx = blockIdx.x * 256 + threadIdx.x;   // 1024*3072 total
    const int d = idx / 3072, c = idx % 3072;
    const float* W; int cc;
    if (c < 1024)      { W = Wq; cc = c; }
    else if (c < 2048) { W = Wk; cc = c - 1024; }
    else               { W = Wv; cc = c - 2048; }
    const int h = cc >> 6, k = cc & 63;
    Wqkv[(size_t)d * 3072 + c] = tobf(W[((size_t)h * 1024 + d) * 64 + k]);
}

__global__ __launch_bounds__(256) void f32_to_bf16_kernel(
    const float* __restrict__ in, __bf16* __restrict__ out, int n)
{
    const int i = blockIdx.x * 256 + threadIdx.x;
    if (i < n) out[i] = tobf(in[i]);
}

// ---------------------------------------------------------------- generic WMMA GEMM core
// C[M,N] = A[M,K] * B[K,N], A/B bf16 row-major, f32 accum, epilogue per element.
// Block: 256 threads = 8 waves (4 M-pos x 2 N-pos); block tile 128x128; wave tile
// 32x64; K-step 64 (2 sub-steps of 32); double-buffered LDS, one barrier per step.
// Per barrier interval each wave issues 16 v_wmma. Fragment loads kept pairwise
// (per-ns) to bound register liveness -- batching them spills to scratch.

template <typename Epi>
__device__ __forceinline__ void gemm_core(const __bf16* __restrict__ A,
                                          const __bf16* __restrict__ Bm,
                                          int K, int N, Epi epi)
{
    __shared__ __align__(32) __bf16 sA[2][128][64];
    __shared__ __align__(32) __bf16 sB[2][64][128];
    const int bm = blockIdx.y * 128;
    const int bn = blockIdx.x * 128;
    const int tid  = threadIdx.x;
    const int wave = tid >> 5, lane = tid & 31;
    const int wm = wave & 3, wn = wave >> 2;          // wave tile origin: (wm*32, wn*64)
    const int half = lane >> 4, l16 = lane & 15;
    const int arow = tid >> 1, acol = (tid & 1) * 32; // A tile: 128x64, 32 elems/thread
    const int brow = tid >> 2, bcol = (tid & 3) * 32; // B tile: 64x128, 32 elems/thread

    v8f acc[2][4];
    #pragma unroll
    for (int ms = 0; ms < 2; ++ms)
        #pragma unroll
        for (int ns = 0; ns < 4; ++ns)
            #pragma unroll
            for (int i = 0; i < 8; ++i) acc[ms][ns][i] = 0.f;

    uint4 ra[4], rb[4];
    const int KT = K >> 6;                            // K-steps of 64

    // prologue: fetch tile 0 into buffer 0
    #pragma unroll
    for (int i = 0; i < 4; ++i) {
        ra[i] = *(const uint4*)(A  + (size_t)(bm + arow) * K + acol + i * 8);
        rb[i] = *(const uint4*)(Bm + (size_t)brow * N + bn + bcol + i * 8);
    }
    #pragma unroll
    for (int i = 0; i < 4; ++i) {
        *(uint4*)&sA[0][arow][acol + i * 8] = ra[i];
        *(uint4*)&sB[0][brow][bcol + i * 8] = rb[i];
    }
    __syncthreads();

    for (int kt = 0; kt < KT; ++kt) {
        const int cur = kt & 1;
        const bool more = (kt + 1) < KT;
        // prefetch tile kt+2 into L2 (global_prefetch_b8)
        if (kt + 2 < KT) {
            const int kp = (kt + 2) << 6;
            __builtin_prefetch(A  + (size_t)(bm + arow) * K + kp + acol, 0, 2);
            __builtin_prefetch(Bm + (size_t)(kp + brow) * N + bn + bcol, 0, 2);
        }
        if (more) {
            const int k0 = (kt + 1) << 6;
            #pragma unroll
            for (int i = 0; i < 4; ++i) {
                ra[i] = *(const uint4*)(A  + (size_t)(bm + arow) * K + k0 + acol + i * 8);
                rb[i] = *(const uint4*)(Bm + (size_t)(k0 + brow) * N + bn + bcol + i * 8);
            }
        }
        #pragma unroll
        for (int ks = 0; ks < 2; ++ks) {
            v16bf a[2];
            #pragma unroll
            for (int ms = 0; ms < 2; ++ms) {
                const __bf16* ap = &sA[cur][wm * 32 + ms * 16 + l16][ks * 32];
                v8bf lo = *(const v8bf*)(ap + half * 8);
                v8bf hi = *(const v8bf*)(ap + 16 + half * 8);
                a[ms] = combine8(lo, hi);
            }
            #pragma unroll
            for (int ns = 0; ns < 4; ++ns) {
                v16bf bfr;
                #pragma unroll
                for (int s = 0; s < 16; ++s)
                    bfr[s] = sB[cur][ks * 32 + half * 8 + (s & 7) + (s >> 3) * 16][wn * 64 + ns * 16 + l16];
                acc[0][ns] = wmma_bf16(a[0], bfr, acc[0][ns]);
                acc[1][ns] = wmma_bf16(a[1], bfr, acc[1][ns]);
            }
        }
        if (more) {
            const int nxt = cur ^ 1;
            #pragma unroll
            for (int i = 0; i < 4; ++i) {
                *(uint4*)&sA[nxt][arow][acol + i * 8] = ra[i];
                *(uint4*)&sB[nxt][brow][bcol + i * 8] = rb[i];
            }
        }
        __syncthreads();
    }

    // C layout: VGPR r -> row r (lanes 0-15) / row r+8 (lanes 16-31), col = lane%16
    #pragma unroll
    for (int ms = 0; ms < 2; ++ms)
        #pragma unroll
        for (int ns = 0; ns < 4; ++ns)
            #pragma unroll
            for (int r = 0; r < 8; ++r) {
                const int row = bm + wm * 32 + ms * 16 + r + half * 8;
                const int col = bn + wn * 64 + ns * 16 + l16;
                epi(row, col, acc[ms][ns][r]);
            }
}

// ---------------------------------------------------------------- GEMM kernels w/ fused epilogues

__global__ __launch_bounds__(256) void gemm_qkv_kernel(
    const __bf16* __restrict__ hbf, const __bf16* __restrict__ Wqkv,
    const float* __restrict__ bq, const float* __restrict__ bk, const float* __restrict__ bv,
    __bf16* __restrict__ Qb, __bf16* __restrict__ Ktb, __bf16* __restrict__ Vb)
{
    auto epi = [=](int row, int col, float v) {
        const int b = row >> 11, s = row & 2047;
        if (col < 1024) {            // Q [B,H,S,DK], fold 1/sqrt(64)
            const int h = col >> 6, k = col & 63;
            Qb[(((size_t)(b * 16 + h)) * 2048 + s) * 64 + k] = tobf((v + bq[col]) * 0.125f);
        } else if (col < 2048) {     // K^T [B,H,DK,S]
            const int cc = col - 1024, h = cc >> 6, k = cc & 63;
            Ktb[(((size_t)(b * 16 + h)) * 64 + k) * 2048 + s] = tobf(v + bk[cc]);
        } else {                     // V [B,H,S,DV]
            const int cc = col - 2048, h = cc >> 6, k = cc & 63;
            Vb[(((size_t)(b * 16 + h)) * 2048 + s) * 64 + k] = tobf(v + bv[cc]);
        }
    };
    gemm_core(hbf, Wqkv, 1024, 3072, epi);
}

__global__ __launch_bounds__(256) void gemm_wo_kernel(
    const __bf16* __restrict__ Zc, const __bf16* __restrict__ Wo,
    const float* __restrict__ bo, const float* __restrict__ x, float* __restrict__ x2)
{
    auto epi = [=](int row, int col, float v) {
        const size_t idx = (size_t)row * 1024 + col;
        x2[idx] = x[idx] + v + bo[col];
    };
    gemm_core(Zc, Wo, 1024, 1024, epi);
}

__global__ __launch_bounds__(256) void gemm_ff1_kernel(
    const __bf16* __restrict__ h2, const __bf16* __restrict__ W1b,
    const float* __restrict__ b1, __bf16* __restrict__ t1)
{
    auto epi = [=](int row, int col, float v) {
        t1[(size_t)row * 4096 + col] = tobf(fmaxf(v + b1[col], 0.0f));
    };
    gemm_core(h2, W1b, 1024, 4096, epi);
}

__global__ __launch_bounds__(256) void gemm_ff2_kernel(
    const __bf16* __restrict__ t1, const __bf16* __restrict__ W2b,
    const float* __restrict__ b2, const float* __restrict__ x2, float* __restrict__ outp)
{
    auto epi = [=](int row, int col, float v) {
        const size_t idx = (size_t)row * 1024 + col;
        outp[idx] = x2[idx] + v + b2[col];
    };
    gemm_core(t1, W2b, 4096, 1024, epi);
}

// ---------------------------------------------------------------- flash attention
// grid (S/64, H, B), 128 threads = 4 waves, one 16-query tile per wave.

__global__ __launch_bounds__(128) void attn_kernel(
    const __bf16* __restrict__ Q, const __bf16* __restrict__ Kt,
    const __bf16* __restrict__ V, __bf16* __restrict__ Zc)
{
    const int S = 2048, DK = 64;
    const int wave = threadIdx.x >> 5, lane = threadIdx.x & 31;
    const int half = lane >> 4, l16 = lane & 15;
    const int b = blockIdx.z, h = blockIdx.y;
    const int s0 = blockIdx.x * 64 + wave * 16;
    const int bh = b * 16 + h;
    const __bf16* Qp = Q  + ((size_t)bh * S + s0) * DK;
    const __bf16* Kp = Kt + (size_t)bh * DK * S;
    const __bf16* Vp = V  + (size_t)bh * S * DK;

    __shared__ __align__(32) __bf16 sP[4][16][32];   // per-wave P staging

    // Q A-fragments for K-dim 0..31 and 32..63 (loaded once)
    v16bf qa[2];
    {
        const __bf16* qrow = Qp + (size_t)l16 * DK;
        #pragma unroll
        for (int kk = 0; kk < 2; ++kk) {
            v8bf lo = *(const v8bf*)(qrow + kk * 32 + half * 8);
            v8bf hi = *(const v8bf*)(qrow + kk * 32 + 16 + half * 8);
            qa[kk] = combine8(lo, hi);
        }
    }

    float m[8], l[8];
    v8f acc[4];
    #pragma unroll
    for (int r = 0; r < 8; ++r) { m[r] = -3.0e38f; l[r] = 0.f; }
    #pragma unroll
    for (int nb = 0; nb < 4; ++nb)
        #pragma unroll
        for (int i = 0; i < 8; ++i) acc[nb][i] = 0.f;

    for (int t0 = 0; t0 < S; t0 += 32) {
        // prefetch next K/V chunk into L2
        if (t0 + 32 < S) {
            __builtin_prefetch(Kp + (size_t)lane * S + t0 + 32, 0, 2);
            __builtin_prefetch(Kp + (size_t)(32 + lane) * S + t0 + 32, 0, 2);
            __builtin_prefetch(Vp + (size_t)(t0 + 32 + lane) * DK, 0, 2);
        }
        // scores S = Q K^T  (16 x 32): batch 4 B-frag loads, then 4 WMMAs
        v16bf bk[4];
        #pragma unroll
        for (int kk = 0; kk < 2; ++kk) {
            #pragma unroll
            for (int s = 0; s < 16; ++s) {
                const __bf16* krow = Kp + (size_t)(kk * 32 + half * 8 + (s & 7) + (s >> 3) * 16) * S + t0;
                bk[kk * 2][s]     = krow[l16];
                bk[kk * 2 + 1][s] = krow[16 + l16];
            }
        }
        v8f sc0, sc1;
        #pragma unroll
        for (int i = 0; i < 8; ++i) { sc0[i] = 0.f; sc1[i] = 0.f; }
        sc0 = wmma_bf16(qa[0], bk[0], sc0);
        sc1 = wmma_bf16(qa[0], bk[1], sc1);
        sc0 = wmma_bf16(qa[1], bk[2], sc0);
        sc1 = wmma_bf16(qa[1], bk[3], sc1);
        // online softmax (row-wise: reduce across the 16 lanes of each half)
        float scale[8];
        #pragma unroll
        for (int r = 0; r < 8; ++r) {
            float mx = fmaxf(sc0[r], sc1[r]);
            #pragma unroll
            for (int off = 8; off >= 1; off >>= 1) mx = fmaxf(mx, __shfl_xor(mx, off, 32));
            const float mnew = fmaxf(m[r], mx);
            const float scl  = __expf(m[r] - mnew);
            const float p0 = __expf(sc0[r] - mnew);
            const float p1 = __expf(sc1[r] - mnew);
            float rsum = p0 + p1;
            #pragma unroll
            for (int off = 8; off >= 1; off >>= 1) rsum += __shfl_xor(rsum, off, 32);
            l[r] = l[r] * scl + rsum;
            m[r] = mnew;
            scale[r] = scl;
            sc0[r] = p0; sc1[r] = p1;
        }
        #pragma unroll
        for (int nb = 0; nb < 4; ++nb)
            #pragma unroll
            for (int r = 0; r < 8; ++r) acc[nb][r] *= scale[r];
        // stage P (C layout -> LDS -> A layout); same-wave LDS ops are in order
        #pragma unroll
        for (int r = 0; r < 8; ++r) {
            sP[wave][r + half * 8][l16]      = tobf(sc0[r]);
            sP[wave][r + half * 8][16 + l16] = tobf(sc1[r]);
        }
        v8bf plo = *(const v8bf*)&sP[wave][l16][half * 8];
        v8bf phi = *(const v8bf*)&sP[wave][l16][16 + half * 8];
        v16bf pa = combine8(plo, phi);
        // O += P V  (K-dim = 32 t-chunk): batch 4 V-frag loads, then 4 WMMAs
        v16bf bv[4];
        #pragma unroll
        for (int nb = 0; nb < 4; ++nb)
            #pragma unroll
            for (int s = 0; s < 16; ++s)
                bv[nb][s] = Vp[(size_t)(t0 + half * 8 + (s & 7) + (s >> 3) * 16) * DK + nb * 16 + l16];
        #pragma unroll
        for (int nb = 0; nb < 4; ++nb)
            acc[nb] = wmma_bf16(pa, bv[nb], acc[nb]);
    }
    // normalize and write Z in concat layout [B, S, H*DV]
    #pragma unroll
    for (int r = 0; r < 8; ++r) {
        const float invl = 1.0f / l[r];
        const int row = s0 + r + half * 8;
        #pragma unroll
        for (int nb = 0; nb < 4; ++nb) {
            const int col = h * 64 + nb * 16 + l16;
            Zc[((size_t)b * S + row) * 1024 + col] = tobf(acc[nb][r] * invl);
        }
    }
}

// ---------------------------------------------------------------- launch

extern "C" void kernel_launch(void* const* d_in, const int* in_sizes, int n_in,
                              void* d_out, int out_size, void* d_ws, size_t ws_size,
                              hipStream_t stream) {
    (void)in_sizes; (void)n_in; (void)out_size; (void)ws_size;
    const float* x     = (const float*)d_in[0];
    const float* Wq    = (const float*)d_in[1];
    const float* bq    = (const float*)d_in[2];
    const float* Wk    = (const float*)d_in[3];
    const float* bk    = (const float*)d_in[4];
    const float* Wv    = (const float*)d_in[5];
    const float* bv    = (const float*)d_in[6];
    const float* Wo    = (const float*)d_in[7];
    const float* bo    = (const float*)d_in[8];
    const float* W1    = (const float*)d_in[9];
    const float* b1    = (const float*)d_in[10];
    const float* W2    = (const float*)d_in[11];
    const float* b2    = (const float*)d_in[12];
    const float* ln1_g = (const float*)d_in[13];
    const float* ln1_b = (const float*)d_in[14];
    const float* ln2_g = (const float*)d_in[15];
    const float* ln2_b = (const float*)d_in[16];
    float* outp = (float*)d_out;

    char* ws = (char*)d_ws;
    __bf16* hbf   = (__bf16*)(ws + 0);           // 16 MB  (reused for h2)
    __bf16* Wqkvb = (__bf16*)(ws + 16777216);    //  6 MB
    __bf16* Wob   = (__bf16*)(ws + 23068672);    //  2 MB
    __bf16* W1b   = (__bf16*)(ws + 25165824);    //  8 MB
    __bf16* W2b   = (__bf16*)(ws + 33554432);    //  8 MB
    __bf16* Qb    = (__bf16*)(ws + 41943040);    // 16 MB
    __bf16* Ktb   = (__bf16*)(ws + 58720256);    // 16 MB
    __bf16* Vb    = (__bf16*)(ws + 75497472);    // 16 MB
    __bf16* Zc    = (__bf16*)(ws + 92274688);    // 16 MB
    float*  x2    = (float*) (ws + 109051904);   // 32 MB
    __bf16* t1    = (__bf16*)(ws + 142606336);   // 64 MB -> total 200 MB

    // 1. LN1 -> bf16
    ln_bf16_kernel<<<8192, 256, 0, stream>>>(x, ln1_g, ln1_b, hbf);
    // 2. weight conversion
    convert_qkv_w<<<(1024 * 3072) / 256, 256, 0, stream>>>(Wq, Wk, Wv, Wqkvb);
    f32_to_bf16_kernel<<<(1024 * 1024) / 256, 256, 0, stream>>>(Wo, Wob, 1024 * 1024);
    f32_to_bf16_kernel<<<(1024 * 4096) / 256, 256, 0, stream>>>(W1, W1b, 1024 * 4096);
    f32_to_bf16_kernel<<<(4096 * 1024) / 256, 256, 0, stream>>>(W2, W2b, 4096 * 1024);
    // 3. fused QKV projection (M=8192, N=3072, K=1024)
    gemm_qkv_kernel<<<dim3(3072 / 128, 8192 / 128), 256, 0, stream>>>(
        hbf, Wqkvb, bq, bk, bv, Qb, Ktb, Vb);
    // 4. flash attention
    attn_kernel<<<dim3(2048 / 64, 16, 4), 128, 0, stream>>>(Qb, Ktb, Vb, Zc);
    // 5. output projection + residual -> x2
    gemm_wo_kernel<<<dim3(1024 / 128, 8192 / 128), 256, 0, stream>>>(Zc, Wob, bo, x, x2);
    // 6. LN2 -> bf16 (reuse hbf)
    ln_bf16_kernel<<<8192, 256, 0, stream>>>(x2, ln2_g, ln2_b, hbf);
    // 7. FFN1 + ReLU
    gemm_ff1_kernel<<<dim3(4096 / 128, 8192 / 128), 256, 0, stream>>>(hbf, W1b, b1, t1);
    // 8. FFN2 + residual -> out
    gemm_ff2_kernel<<<dim3(1024 / 128, 8192 / 128), 256, 0, stream>>>(t1, W2b, b2, x2, outp);
}
